// KNNLayer_79010218377296
// MI455X (gfx1250) — matrix operand order repaired and checked by reading, not compiled, not run on hardware
//
#include <hip/hip_runtime.h>
#include <hip/hip_bf16.h>
#include <math.h>

typedef __bf16 v16bf __attribute__((ext_vector_type(16)));
typedef float  v8f   __attribute__((ext_vector_type(8)));

#define Bn 32
#define Nn 1024
#define Fn 128
#define Pn 128
#define Kn 16

// ---------------------------------------------------------------------------
// gelu via gfx1250 V_TANH_F32 (TRANS op). tanh-form gelu deviates from exact
// erf-gelu by <~1e-3 abs, below the bf16 quantization noise of the WMMA path.
// A v_nop inside the asm covers the TRANS-result hazard (1 indep op needed).
// ---------------------------------------------------------------------------
__device__ __forceinline__ float gelu_tanh(float x) {
    float inner = x * (0.79788456080286535f + 0.035677408136300125f * x * x);
    float t;
    asm volatile("v_tanh_f32 %0, %1\n\tv_nop" : "=v"(t) : "v"(inner));
    return 0.5f * x * (1.0f + t);
}

// ---------------------------------------------------------------------------
// Kernel 1: pairwise sq-distance + top-16 neighbor selection (self excluded).
// One thread per query point; batch's points staged in LDS (8 KB). Neighbor
// order is irrelevant downstream (mean over K), so insertion-select == the
// reference top_k(K+1)[1:] as a set.
// ---------------------------------------------------------------------------
__global__ __launch_bounds__(256) void topk_kernel(const float* __restrict__ points,
                                                   int* __restrict__ idx_out) {
    __shared__ float pts[Nn * 2];
    const int b   = blockIdx.y;
    const int tid = threadIdx.x;
    const float* pb = points + (size_t)b * Nn * 2;
    for (int e = tid; e < Nn * 2; e += 256) pts[e] = pb[e];
    __syncthreads();

    const int n = blockIdx.x * 256 + tid;
    const float px = pts[2 * n], py = pts[2 * n + 1];

    float dist[Kn];
    int   ind[Kn];
#pragma unroll
    for (int s = 0; s < Kn; ++s) { dist[s] = 3.4e38f; ind[s] = 0; }

    for (int m = 0; m < Nn; ++m) {
        float dx = px - pts[2 * m];
        float dy = py - pts[2 * m + 1];
        float d  = dx * dx + dy * dy;
        if (m == n) continue;
        if (d < dist[Kn - 1]) {
            dist[Kn - 1] = d; ind[Kn - 1] = m;
#pragma unroll
            for (int s = Kn - 1; s > 0; --s) {
                if (dist[s] < dist[s - 1]) {
                    float td = dist[s]; dist[s] = dist[s - 1]; dist[s - 1] = td;
                    int   ti = ind[s];  ind[s]  = ind[s - 1];  ind[s - 1]  = ti;
                }
            }
        }
    }
    int* op = idx_out + ((size_t)b * Nn + n) * Kn;
#pragma unroll
    for (int s = 0; s < Kn; ++s) op[s] = ind[s];
}

// ---------------------------------------------------------------------------
// Kernel 2: fused gather + GEMM1(16x256 @ 256x256) + gelu + GEMM2(16x256 @
// 256x128) + gelu + mean-over-K in bf16 WMMA (f32 accumulate).
//
// Each wave owns TWO n's so every B-fragment ds_load feeds two v_wmma ops
// (2x arithmetic intensity on LDS, paired wmmas hide the dscnt waits).
//
// LDS map (dynamic, 327680 B == the 320 KB WGP maximum):
//   w1f : 65536 bf16 (B-frag layout) @ 0
//   w2f : 32768 bf16 (B-frag layout) @ 131072
//   hst : 8 waves x 2 x 4096 bf16    @ 196608   (A-frag layout for GEMM2)
//
// B-fragment (tile j, chunk k): element(lane l, slot i) =
//   W[row = k*32 + (i>>3)*16 + (l>>4)*8 + (i&7)][col = j*16 + (l&15)]
// A 16x32 bf16: lane l -> M = l&15 ; slot i (pair p=i>>1) covers K-col
//   off = ((p>>2)<<4) + ((p&3)<<1) + ((l>=16)?8:0) + (i&1)
// C/D f32: reg r, lane l -> M = r + 8*(l>>4), N = l&15
// ---------------------------------------------------------------------------
__global__ __launch_bounds__(256) void mlp_kernel(const float* __restrict__ features,
                                                  const float* __restrict__ W1,
                                                  const float* __restrict__ b1,
                                                  const float* __restrict__ W2,
                                                  const float* __restrict__ b2,
                                                  const int*   __restrict__ idx,
                                                  float* __restrict__ out) {
    extern __shared__ char smem[];
    __bf16* w1f = (__bf16*)smem;                  // 65536 bf16
    __bf16* w2f = w1f + 65536;                    // 32768 bf16
    __bf16* hst = w2f + 32768;                    // 8 waves * 8192 bf16

    const int tid = threadIdx.x;
    const int b   = blockIdx.y;

    // Stage W1/W2 into bf16 B-fragment layout (linear LDS index e == frag idx).
    for (int e = tid; e < 256 * 256; e += 256) {
        int i = e & 15, l = (e >> 4) & 31, f = e >> 9;
        int k = f & 7, j = f >> 3;
        int row = k * 32 + ((i >> 3) << 4) + ((l >> 4) << 3) + (i & 7);
        int col = j * 16 + (l & 15);
        w1f[e] = (__bf16)W1[row * 256 + col];
    }
    for (int e = tid; e < 256 * 128; e += 256) {
        int i = e & 15, l = (e >> 4) & 31, f = e >> 9;
        int k = f & 7, j = f >> 3;
        int row = k * 32 + ((i >> 3) << 4) + ((l >> 4) << 3) + (i & 7);
        int col = j * 16 + (l & 15);
        w2f[e] = (__bf16)W2[row * 128 + col];
    }
    __syncthreads();

    const int wave = tid >> 5;
    const int l    = tid & 31;
    const int lrow = l & 15;     // A-row (neighbor id) == D-column-in-tile
    const int lhi  = l >> 4;
    __bf16* hw = hst + wave * 8192;        // two 4096-elem buffers per wave

    for (int it = 0; it < 4; ++it) {
        const int nbase = blockIdx.x * 64 + it * 16 + (wave << 1);

        // Prefetch next iteration's center rows (global_prefetch_b8).
        if (it < 3) {
            __builtin_prefetch(features + ((size_t)b * Nn + nbase + 16) * Fn, 0, 1);
            __builtin_prefetch(features + ((size_t)b * Nn + nbase + 17) * Fn, 0, 1);
        }

        // Build A1 fragments for local = [knn - center, center] for both n's.
        v16bf a1[2][8];
#pragma unroll
        for (int t = 0; t < 2; ++t) {
            const int n   = nbase + t;
            const int nbr = idx[((size_t)b * Nn + n) * Kn + lrow];
            const float* cf = features + ((size_t)b * Nn + n)   * Fn;
            const float* nf = features + ((size_t)b * Nn + nbr) * Fn;
#pragma unroll
            for (int k = 0; k < 8; ++k) {
#pragma unroll
                for (int p = 0; p < 8; ++p) {
                    int off = ((p >> 2) << 4) + ((p & 3) << 1) + (lhi << 3);
                    int c   = k * 32 + off;
                    float v0, v1;
                    if (c < 128) { v0 = nf[c] - cf[c]; v1 = nf[c + 1] - cf[c + 1]; }
                    else         { v0 = cf[c - 128];   v1 = cf[c - 127]; }
                    a1[t][k][2 * p]     = (__bf16)v0;
                    a1[t][k][2 * p + 1] = (__bf16)v1;
                }
            }
        }

        // GEMM1 + gelu -> per-wave h staging (A2 fragment layout).
#pragma unroll 1
        for (int j = 0; j < 16; ++j) {
            float bias = b1[j * 16 + lrow];       // L2-resident
            v8f cc[2];
#pragma unroll
            for (int r = 0; r < 8; ++r) { cc[0][r] = bias; cc[1][r] = bias; }
#pragma unroll
            for (int k = 0; k < 8; ++k) {
                v16bf bf = *(const v16bf*)(w1f + ((j * 8 + k) * 32 + l) * 16);
                cc[0] = __builtin_amdgcn_wmma_f32_16x16x32_bf16(false, a1[0][k], false, bf,
                                                                (short)0, cc[0], false, false);
                cc[1] = __builtin_amdgcn_wmma_f32_16x16x32_bf16(false, a1[1][k], false, bf,
                                                                (short)0, cc[1], false, false);
            }
            // D(lane l, reg r) -> h[M = r + 8*lhi][ch = j*16 + lrow], stored in
            // A-fragment addressing for GEMM2.
            int ch   = j * 16 + lrow;
            int k2   = ch >> 5;
            int off  = ch & 31;
            int i2   = (((((off >> 4) << 2) + ((off & 7) >> 1)) << 1) | (off & 1));
            int lsel = ((off & 15) >= 8) ? 16 : 0;
#pragma unroll
            for (int t = 0; t < 2; ++t) {
#pragma unroll
                for (int r = 0; r < 8; ++r) {
                    int row = r + (lhi << 3);
                    float g = gelu_tanh(cc[t][r]);
                    hw[t * 4096 + (k2 * 32 + (row + lsel)) * 16 + i2] = (__bf16)g;
                }
            }
        }

        // GEMM2 + gelu + mean over the 16 neighbor rows, for both n's.
#pragma unroll 1
        for (int j = 0; j < 8; ++j) {
            float bias = b2[j * 16 + lrow];
            v8f cc[2];
#pragma unroll
            for (int r = 0; r < 8; ++r) { cc[0][r] = bias; cc[1][r] = bias; }
#pragma unroll
            for (int k = 0; k < 8; ++k) {
                v16bf bf = *(const v16bf*)(w2f + ((j * 8 + k) * 32 + l) * 16);
#pragma unroll
                for (int t = 0; t < 2; ++t) {
                    v16bf av = *(const v16bf*)(hw + t * 4096 + (k * 32 + l) * 16);
                    cc[t] = __builtin_amdgcn_wmma_f32_16x16x32_bf16(false, av, false, bf,
                                                                    (short)0, cc[t], false, false);
                }
            }
#pragma unroll
            for (int t = 0; t < 2; ++t) {
                float s = 0.0f;
#pragma unroll
                for (int r = 0; r < 8; ++r) s += gelu_tanh(cc[t][r]);
                s += __shfl_xor(s, 16, 32);       // combine M=0..7 with M=8..15
                if (l < 16)
                    out[((size_t)b * Nn + nbase + t) * Pn + j * 16 + l] = s * (1.0f / 16.0f);
            }
        }
    }
}

extern "C" void kernel_launch(void* const* d_in, const int* in_sizes, int n_in,
                              void* d_out, int out_size, void* d_ws, size_t ws_size,
                              hipStream_t stream) {
    const float* points   = (const float*)d_in[0];
    const float* features = (const float*)d_in[1];
    const float* W1       = (const float*)d_in[2];
    const float* b1       = (const float*)d_in[3];
    const float* W2       = (const float*)d_in[4];
    const float* b2       = (const float*)d_in[5];
    float* out  = (float*)d_out;
    int*   idxb = (int*)d_ws;   // B*N*K ints = 2 MB scratch

    topk_kernel<<<dim3(Nn / 256, Bn), 256, 0, stream>>>(points, idxb);

    const size_t lds_bytes = (65536 + 32768 + 8 * 8192) * 2;   // 327680 = 320 KB
    mlp_kernel<<<dim3(16, Bn), 256, lds_bytes, stream>>>(features, W1, b1, W2, b2,
                                                         idxb, out);
}
